// GNNSelector_17738214933181
// MI455X (gfx1250) — matrix operand order repaired and checked by reading, not compiled, not run on hardware
//
#include <hip/hip_runtime.h>

#define IN_CH   256
#define HIDC    64
#define HEADS1  2
#define NEGS    0.2f
#define BNEPS   1e-5f
#define KEY_NEGINF 0x007FFFFFu   // key(-inf) under monotonic float->uint map
#define CDIV(a,b) (((a)+(b)-1)/(b))

typedef __bf16 v16bf __attribute__((ext_vector_type(16)));
typedef float  v8f   __attribute__((ext_vector_type(8)));

union ABf { v16bf v; unsigned int u[8]; };

__device__ __forceinline__ unsigned short f2bf(float f) {
  unsigned u = __float_as_uint(f);
  u += 0x7FFFu + ((u >> 16) & 1u);          // round-to-nearest-even
  return (unsigned short)(u >> 16);
}

// monotonic float -> uint key (order-preserving), and inverse
__device__ __forceinline__ unsigned f2key(float f) {
  unsigned iv = __float_as_uint(f);
  return (iv & 0x80000000u) ? ~iv : (iv | 0x80000000u);
}
__device__ __forceinline__ float key2f(unsigned k) {
  unsigned iv = (k & 0x80000000u) ? (k & 0x7FFFFFFFu) : ~k;
  return __uint_as_float(iv);
}

__device__ __forceinline__ void edge_sd(const int* __restrict__ ei, int e, int E0,
                                        int& s, int& d) {
  if (e < E0) { s = ei[e]; d = ei[E0 + e]; }
  else        { s = e - E0; d = s; }        // self loops appended
}

// ---------------- utility kernels ----------------

__global__ void k_fill_u32(unsigned* __restrict__ p, unsigned v, int n) {
  int t = blockIdx.x * blockDim.x + threadIdx.x;
  if (t < n) p[t] = v;
}

__global__ void k_cast_bf16(const float* __restrict__ in,
                            unsigned short* __restrict__ out, int n) {
  int t = blockIdx.x * blockDim.x + threadIdx.x;
  if (t < n) out[t] = f2bf(in[t]);
}

// W [K, Nc] f32 row-major  ->  Wt [Nc, K] bf16 (K contiguous per output row)
__global__ void k_castT_bf16(const float* __restrict__ W,
                             unsigned short* __restrict__ Wt, int K, int Nc) {
  int t = blockIdx.x * blockDim.x + threadIdx.x;
  if (t >= K * Nc) return;
  int k = t / Nc, n = t - k * Nc;
  Wt[n * K + k] = f2bf(W[t]);
}

// ---------------- WMMA GEMM: C[M,N] = A[M,K](bf16) * B[K,N] + bias ----------------
// Bt is stored transposed [N,K] bf16. One wave per 16x16 tile. M,N,K % 16|32 == 0.
__global__ __launch_bounds__(256) void k_wmma_gemm(
    const unsigned short* __restrict__ A,
    const unsigned short* __restrict__ Bt,
    const float* __restrict__ bias,
    float* __restrict__ C,
    int M, int K, int N) {
  int lane  = threadIdx.x & 31;
  int wave  = threadIdx.x >> 5;
  int gw    = blockIdx.x * 8 + wave;
  int ntN   = N >> 4;
  int tm    = gw / ntN;
  int tn    = gw - tm * ntN;
  if (tm * 16 >= M) return;                 // wave-uniform: EXEC stays all-1s

  int col16 = lane & 15;
  int hi    = lane >> 4;

  const unsigned short* arow = A  + (size_t)(tm * 16 + col16) * K;
  const unsigned short* brow = Bt + (size_t)(tn * 16 + col16) * K;

  v8f acc = {0.f, 0.f, 0.f, 0.f, 0.f, 0.f, 0.f, 0.f};

  for (int k = 0; k < K; k += 32) {
    ABf a, b;
#pragma unroll
    for (int j = 0; j < 8; ++j) {
      // A 16x32 bf16 layout: lanes<16 hold K {0..7,16..23}, lanes>=16 hold K {8..15,24..31}
      int ka = k + ((j < 4) ? (2 * j) : (2 * j + 8)) + hi * 8;
      a.u[j] = *(const unsigned int*)(arow + ka);
      // B 32x16 bf16 layout: lanes<16 hold K {0..7,16..23}, lanes>=16 hold K {8..15,24..31}
      int kb = k + ((j < 4) ? (2 * j) : (2 * j + 8)) + hi * 8;
      b.u[j] = *(const unsigned int*)(brow + kb);
    }
    acc = __builtin_amdgcn_wmma_f32_16x16x32_bf16(
        false, a.v, false, b.v, (short)0, acc, false, false);
  }

  int colg = tn * 16 + col16;
  float bv = bias[colg];
#pragma unroll
  for (int r = 0; r < 8; ++r) {
    int row = tm * 16 + hi * 8 + r;         // C layout: VGPR r -> M = r (+8 for lanes>=16)
    C[(size_t)row * N + colg] = acc[r] + bv;
  }
}

// ---------------- edge kernels ----------------

// logits[e,h] = att[h,:] . leaky_relu(XL[src,h,:] + XR[dst,h,:]); atomic max into mkey[dst,h]
__global__ void k_logits(const int* __restrict__ ei, int E0, int ET, int H,
                         const float* __restrict__ XL, const float* __restrict__ XR,
                         const float* __restrict__ att,
                         float* __restrict__ logits, unsigned* __restrict__ mkey) {
  int t = blockIdx.x * blockDim.x + threadIdx.x;
  if (t >= ET * H) return;
  int e = t / H, h = t - e * H;
  int sN, dN; edge_sd(ei, e, E0, sN, dN);
  const float* xl = XL + (size_t)(sN * H + h) * HIDC;
  const float* xr = XR + (size_t)(dN * H + h) * HIDC;
  const float* at = att + h * HIDC;
  float acc = 0.f;
#pragma unroll 8
  for (int c = 0; c < HIDC; ++c) {
    float v = xl[c] + xr[c];
    v = v > 0.f ? v : NEGS * v;
    acc += at[c] * v;
  }
  logits[t] = acc;
  atomicMax(mkey + dN * H + h, f2key(acc));
}

// p = exp(logit - max); s[dst,h] += p   (p overwrites logits buffer)
__global__ void k_expsum(const int* __restrict__ ei, int E0, int ET, int H,
                         const unsigned* __restrict__ mkey,
                         float* __restrict__ logits, float* __restrict__ s) {
  int t = blockIdx.x * blockDim.x + threadIdx.x;
  if (t >= ET * H) return;
  int e = t / H, h = t - e * H;
  int sN, dN; edge_sd(ei, e, E0, sN, dN);
  float m = key2f(mkey[dN * H + h]);
  float p = __expf(logits[t] - m);
  logits[t] = p;
  atomicAdd(s + dN * H + h, p);
}

// agg[dst, j] += (p[e,h]/s[dst,h]) * XL[src, j],  j in [0, H*64), h = j/64
__global__ void k_scatter(const int* __restrict__ ei, int E0, int ET, int H, int HC,
                          const float* __restrict__ p, const float* __restrict__ s,
                          const float* __restrict__ XL, float* __restrict__ agg) {
  int t = blockIdx.x * blockDim.x + threadIdx.x;
  if (t >= ET * HC) return;
  int e = t / HC, j = t - e * HC;
  int h = j >> 6;                            // C == 64
  int sN, dN; edge_sd(ei, e, E0, sN, dN);
  float alpha = p[e * H + h] / s[dN * H + h];
  atomicAdd(agg + (size_t)dN * HC + j, alpha * XL[(size_t)sN * HC + j]);
}

// h = elu(bn(agg + bias)); store as bf16 for next WMMA GEMM
__global__ void k_bnelu_bf16(const float* __restrict__ agg, const float* __restrict__ bias,
                             const float* __restrict__ g, const float* __restrict__ b,
                             const float* __restrict__ m, const float* __restrict__ v,
                             unsigned short* __restrict__ outbf, int n, int Cdim) {
  int t = blockIdx.x * blockDim.x + threadIdx.x;
  if (t >= n) return;
  int j = t % Cdim;
  float x = agg[t] + bias[j];
  x = (x - m[j]) * rsqrtf(v[j] + BNEPS) * g[j] + b[j];
  x = x > 0.f ? x : (__expf(x) - 1.f);
  outbf[t] = f2bf(x);
}

// out[i] = sigmoid( elu(bn(agg2[i,:] + bias2)) . Wc + bc )
__global__ void k_final(const float* __restrict__ agg2, const float* __restrict__ bias2,
                        const float* __restrict__ g, const float* __restrict__ b,
                        const float* __restrict__ m, const float* __restrict__ v,
                        const float* __restrict__ Wc, const float* __restrict__ bc,
                        float* __restrict__ out, int n) {
  int i = blockIdx.x * blockDim.x + threadIdx.x;
  if (i >= n) return;
  float acc = bc[0];
#pragma unroll 8
  for (int c = 0; c < HIDC; ++c) {
    float x = agg2[(size_t)i * HIDC + c] + bias2[c];
    x = (x - m[c]) * rsqrtf(v[c] + BNEPS) * g[c] + b[c];
    x = x > 0.f ? x : (__expf(x) - 1.f);
    acc += x * Wc[c];
  }
  out[i] = 1.f / (1.f + __expf(-acc));
}

// ---------------- host orchestration ----------------

extern "C" void kernel_launch(void* const* d_in, const int* in_sizes, int n_in,
                              void* d_out, int out_size, void* d_ws, size_t ws_size,
                              hipStream_t stream) {
  const float* x    = (const float*)d_in[0];
  const int*   ei   = (const int*)d_in[1];
  const float* W1l  = (const float*)d_in[2];
  const float* b1l  = (const float*)d_in[3];
  const float* W1r  = (const float*)d_in[4];
  const float* b1r  = (const float*)d_in[5];
  const float* att1 = (const float*)d_in[6];
  const float* bias1= (const float*)d_in[7];
  const float* bn1g = (const float*)d_in[8];
  const float* bn1b = (const float*)d_in[9];
  const float* bn1m = (const float*)d_in[10];
  const float* bn1v = (const float*)d_in[11];
  const float* W2l  = (const float*)d_in[12];
  const float* b2l  = (const float*)d_in[13];
  const float* W2r  = (const float*)d_in[14];
  const float* b2r  = (const float*)d_in[15];
  const float* att2 = (const float*)d_in[16];
  const float* bias2= (const float*)d_in[17];
  const float* bn2g = (const float*)d_in[18];
  const float* bn2b = (const float*)d_in[19];
  const float* bn2m = (const float*)d_in[20];
  const float* bn2v = (const float*)d_in[21];
  const float* Wc   = (const float*)d_in[22];
  const float* bc   = (const float*)d_in[23];
  float* out = (float*)d_out;
  (void)n_in; (void)out_size; (void)ws_size;

  const int Nn = in_sizes[0] / IN_CH;        // 50000 (multiple of 16)
  const int E0 = in_sizes[1] / 2;            // 800000
  const int ET = E0 + Nn;                    // 850000 (with self loops)
  const int F1 = HEADS1 * HIDC;              // 128

  // workspace carve-out
  char* wp = (char*)d_ws;
  auto alloc = [&](size_t bytes) -> void* {
    void* p = (void*)wp;
    wp += (bytes + 255) & ~(size_t)255;
    return p;
  };
  unsigned short* xbf  = (unsigned short*)alloc((size_t)Nn * IN_CH * 2);
  unsigned short* W1lt = (unsigned short*)alloc((size_t)IN_CH * F1 * 2);
  unsigned short* W1rt = (unsigned short*)alloc((size_t)IN_CH * F1 * 2);
  unsigned short* W2lt = (unsigned short*)alloc((size_t)F1 * HIDC * 2);
  unsigned short* W2rt = (unsigned short*)alloc((size_t)F1 * HIDC * 2);
  float*    XL1 = (float*)alloc((size_t)Nn * F1 * 4);
  float*    XR1 = (float*)alloc((size_t)Nn * F1 * 4);
  float*    lg1 = (float*)alloc((size_t)ET * HEADS1 * 4);
  unsigned* m1  = (unsigned*)alloc((size_t)Nn * HEADS1 * 4);
  float*    s1  = (float*)alloc((size_t)Nn * HEADS1 * 4);
  float*    ag1 = (float*)alloc((size_t)Nn * F1 * 4);
  unsigned short* h1bf = (unsigned short*)alloc((size_t)Nn * F1 * 2);
  float*    XL2 = (float*)alloc((size_t)Nn * HIDC * 4);
  float*    XR2 = (float*)alloc((size_t)Nn * HIDC * 4);
  float*    lg2 = (float*)alloc((size_t)ET * 4);
  unsigned* m2  = (unsigned*)alloc((size_t)Nn * 4);
  float*    s2  = (float*)alloc((size_t)Nn * 4);
  float*    ag2 = (float*)alloc((size_t)Nn * HIDC * 4);

  const int TB = 256;

  // casts
  k_cast_bf16<<<CDIV(Nn * IN_CH, TB), TB, 0, stream>>>(x, xbf, Nn * IN_CH);
  k_castT_bf16<<<CDIV(IN_CH * F1, TB), TB, 0, stream>>>(W1l, W1lt, IN_CH, F1);
  k_castT_bf16<<<CDIV(IN_CH * F1, TB), TB, 0, stream>>>(W1r, W1rt, IN_CH, F1);
  k_castT_bf16<<<CDIV(F1 * HIDC, TB), TB, 0, stream>>>(W2l, W2lt, F1, HIDC);
  k_castT_bf16<<<CDIV(F1 * HIDC, TB), TB, 0, stream>>>(W2r, W2rt, F1, HIDC);

  // layer 1 dense: XL1 = x@W1l + b1l ; XR1 = x@W1r + b1r
  {
    int waves  = (Nn / 16) * (F1 / 16);
    int blocks = CDIV(waves, 8);
    k_wmma_gemm<<<blocks, 256, 0, stream>>>(xbf, W1lt, b1l, XL1, Nn, IN_CH, F1);
    k_wmma_gemm<<<blocks, 256, 0, stream>>>(xbf, W1rt, b1r, XR1, Nn, IN_CH, F1);
  }

  // layer 1 edge softmax + aggregation
  k_fill_u32<<<CDIV(Nn * HEADS1, TB), TB, 0, stream>>>(m1, KEY_NEGINF, Nn * HEADS1);
  k_fill_u32<<<CDIV(Nn * HEADS1, TB), TB, 0, stream>>>((unsigned*)s1, 0u, Nn * HEADS1);
  k_fill_u32<<<CDIV(Nn * F1, TB), TB, 0, stream>>>((unsigned*)ag1, 0u, Nn * F1);
  k_logits <<<CDIV(ET * HEADS1, TB), TB, 0, stream>>>(ei, E0, ET, HEADS1, XL1, XR1, att1, lg1, m1);
  k_expsum <<<CDIV(ET * HEADS1, TB), TB, 0, stream>>>(ei, E0, ET, HEADS1, m1, lg1, s1);
  k_scatter<<<CDIV(ET * F1, TB), TB, 0, stream>>>(ei, E0, ET, HEADS1, F1, lg1, s1, XL1, ag1);

  // BN1 + ELU -> bf16 activations for layer 2
  k_bnelu_bf16<<<CDIV(Nn * F1, TB), TB, 0, stream>>>(ag1, bias1, bn1g, bn1b, bn1m, bn1v,
                                                     h1bf, Nn * F1, F1);

  // layer 2 dense
  {
    int waves  = (Nn / 16) * (HIDC / 16);
    int blocks = CDIV(waves, 8);
    k_wmma_gemm<<<blocks, 256, 0, stream>>>(h1bf, W2lt, b2l, XL2, Nn, F1, HIDC);
    k_wmma_gemm<<<blocks, 256, 0, stream>>>(h1bf, W2rt, b2r, XR2, Nn, F1, HIDC);
  }

  // layer 2 edge softmax + aggregation (H = 1)
  k_fill_u32<<<CDIV(Nn, TB), TB, 0, stream>>>(m2, KEY_NEGINF, Nn);
  k_fill_u32<<<CDIV(Nn, TB), TB, 0, stream>>>((unsigned*)s2, 0u, Nn);
  k_fill_u32<<<CDIV(Nn * HIDC, TB), TB, 0, stream>>>((unsigned*)ag2, 0u, Nn * HIDC);
  k_logits <<<CDIV(ET, TB), TB, 0, stream>>>(ei, E0, ET, 1, XL2, XR2, att2, lg2, m2);
  k_expsum <<<CDIV(ET, TB), TB, 0, stream>>>(ei, E0, ET, 1, m2, lg2, s2);
  k_scatter<<<CDIV(ET * HIDC, TB), TB, 0, stream>>>(ei, E0, ET, 1, HIDC, lg2, s2, XL2, ag2);

  // BN2 + ELU + classifier + sigmoid
  k_final<<<CDIV(Nn, TB), TB, 0, stream>>>(ag2, bias2, bn2g, bn2b, bn2m, bn2v, Wc, bc, out, Nn);
}
